// NetKNN_34746285424642
// MI455X (gfx1250) — matrix operand order repaired and checked by reading, not compiled, not run on hardware
//
#include <hip/hip_runtime.h>
#include <math.h>

// ---------------------------------------------------------------------------
// CDNA5 (gfx1250) Gaussian-kernel CNN forward pass.
// - Both RBF convs: implicit im2col GEMM on v_wmma_f32_16x16x32_bf16,
//   one wave -> 16Mx32N (two accumulators, shared A fragment), weights
//   pre-swizzled in LDS.
// - fc1: split-K WMMA GEMM; the ~29MB weight stream is moved by the Tensor
//   Data Mover (tensor_load_to_lds + s_wait_tensorcnt) into LDS, one 43KB
//   tile per block shared by 8 waves; partials combined with
//   global_atomic_add_f32.
// - global_prefetch_b8 via __builtin_prefetch on streaming operands.
// ---------------------------------------------------------------------------

typedef __attribute__((ext_vector_type(16))) __bf16 v16bf;
typedef __attribute__((ext_vector_type(8)))  float  v8f;

#define GAMMA_ 0.05f
#define EPS_   1e-5f

static __device__ __forceinline__ unsigned short f2bfu(float f) {
  // round-to-nearest-even fp32 -> bf16 (bit pattern)
  union { float f; unsigned u; } a; a.f = f;
  unsigned u = a.u + 0x7FFFu + ((a.u >> 16) & 1u);
  return (unsigned short)(u >> 16);
}
static __device__ __forceinline__ float bfu2f(unsigned short s) {
  union { unsigned u; float f; } a; a.u = ((unsigned)s) << 16;
  return a.f;
}

union AFrag { v16bf v; unsigned short u[16]; };

// ---------------------------------------------------------------------------
// Tensor Data Mover: async 2D fp32 tile (rows x cols, row stride in elements)
// from global memory into LDS at byte offset lds_off. Descriptor per
// CDNA5 ISA (D# group0 + group1; groups 2/3 unused -> 0).
// ---------------------------------------------------------------------------
#if __has_builtin(__builtin_amdgcn_tensor_load_to_lds)
#define HAVE_TDM 1
typedef __attribute__((ext_vector_type(4))) unsigned int v4u;
typedef __attribute__((ext_vector_type(4))) int          v4i;
typedef __attribute__((ext_vector_type(8))) int          v8i_;

static __device__ __forceinline__ void tdm_load_2d_f32(
    const float* gsrc, unsigned lds_off, int rows, int cols, int row_stride)
{
  unsigned long long ga = (unsigned long long)(size_t)gsrc;
  v4u g0;
  g0[0] = 1u;                                            // count=1 (valid), user
  g0[1] = lds_off;                                       // LDS byte address
  g0[2] = (unsigned)(ga & 0xFFFFFFFFu);                  // global_addr[31:0]
  g0[3] = (unsigned)((ga >> 32) & 0x01FFFFFFu)           // global_addr[56:32]
        | (2u << 30);                                    // type = 2 ("image")
  v8i_ g1;
  g1[0] = (int)(2u << 16);                               // data_size = 4 bytes
  g1[1] = (int)(((unsigned)row_stride & 0xFFFFu) << 16); // tensor_dim0[15:0]
  g1[2] = (int)((((unsigned)row_stride >> 16) & 0xFFFFu) // tensor_dim0[31:16]
        | ((unsigned)rows << 16));                       // tensor_dim1[15:0]
  g1[3] = (int)((unsigned)cols << 16);                   // tile_dim0
  g1[4] = rows;                                          // tile_dim1 (tile_dim2=0)
  g1[5] = row_stride;                                    // tensor_dim0_stride
  g1[6] = 0;
  g1[7] = 0;
  v4i z4 = {0, 0, 0, 0};
#if defined(__clang_major__) && __clang_major__ >= 23
  v8i_ z8 = {0, 0, 0, 0, 0, 0, 0, 0};
  __builtin_amdgcn_tensor_load_to_lds(g0, g1, z4, z4, z8, 0);
#else
  __builtin_amdgcn_tensor_load_to_lds(g0, g1, z4, z4, 0);
#endif
}
#else
#define HAVE_TDM 0
#endif

// ---------------------------------------------------------------------------
// Gaussian conv as implicit GEMM.
//   out[b,n,oy,ox] = exp(-gamma * (||patch||^2 - 2*patch.w[n] + ||w[n]||^2))
// One wave -> 16(M positions) x 32(N channels): one A fragment feeds two
// back-to-back WMMAs against two B tiles staged (pre-swizzled) in LDS.
// Output stored bf16 (values in (0,1]).
// ---------------------------------------------------------------------------
__global__ __launch_bounds__(128) void gconv_wmma(
    const float* __restrict__ in, const float* __restrict__ w,
    unsigned short* __restrict__ out,
    int Cin, int Hin, int Win, int Cout, int Ho, int Wo,
    int K, int ksteps, float gamma)
{
  extern __shared__ unsigned short lds_b[];   // 2 * ksteps * 512 bf16 entries
  __shared__ float lds_wsq[32];

  const int tid   = threadIdx.x;
  const int wave  = tid >> 5;
  const int lane  = tid & 31;
  const int nbase = blockIdx.y * 32;          // two 16-wide N tiles per block
  const int mtile = blockIdx.x * 4 + wave;

  // ---- stage pre-swizzled B fragments (lane%16 = N, elem = K) for both tiles ----
  const int total = ksteps << 10;             // 2*ksteps*512
  for (int e = tid; e < total; e += 128) {
    int i      = e & 15;                      // element (= K, table order)
    int lane_s = (e >> 4) & 31;               // owning lane (lane%16 = N offset)
    int kp     = e >> 9;                      // kstep*2 + pair
    int kstep  = kp >> 1;
    int pair   = kp & 1;
    int k      = (kstep << 5) + i + 8 * ((i >> 3) + (lane_s >> 4));
    int n      = nbase + pair * 16 + (lane_s & 15);
    lds_b[e] = f2bfu((k < K) ? w[n * K + k] : 0.0f);
  }
  if (tid < 32) {                             // ||w[n]||^2 per channel
    int n = nbase + tid;
    float s = 0.f;
    for (int k = 0; k < K; ++k) { float x = w[n * K + k]; s += x * x; }
    lds_wsq[tid] = s;
  }
  __syncthreads();

  // ---- per-lane A row decode ----
  const int hi   = lane >> 4;
  const int mrow = mtile * 16 + (lane & 15);
  const int hw   = Ho * Wo;
  int bN = mrow / hw;
  int rr = mrow - bN * hw;
  int oy = rr / Wo;
  int ox = rr - oy * Wo;
  const float* inb = in + (size_t)bN * Cin * Hin * Win;

  v8f  acc0 = {}, acc1 = {};
  float asum = 0.f;                           // per-lane partial ||patch||^2

  for (int ks = 0; ks < ksteps; ++ks) {
    AFrag a;
    const int kbase = ks << 5;
    #pragma unroll
    for (int i = 0; i < 16; ++i) {
      // A 16-bit layout: lane<16 holds K = i + 8*(i>=8); lane>=16 adds 8.
      int k = kbase + i + 8 * ((i >> 3) + hi);
      float val = 0.f;
      if (k < K) {
        int c  = k / 25;
        int t  = k - c * 25;
        int ky = t / 5;
        int kx = t - ky * 5;
        val = inb[(c * Hin + oy + ky) * Win + ox + kx];
      }
      asum += val * val;
      a.u[i] = f2bfu(val);
    }
    if (ks + 1 < ksteps) {                    // prefetch next kstep's patch line
      int k = ((ks + 1) << 5) + 8 * hi;
      if (k < K) {
        int c = k / 25; int t = k - c * 25; int ky = t / 5; int kx = t - ky * 5;
        __builtin_prefetch(&inb[(c * Hin + oy + ky) * Win + ox + kx], 0, 1);
      }
    }
    const unsigned short* bbase = lds_b + (ks << 10) + lane * 16;
    acc0 = __builtin_amdgcn_wmma_f32_16x16x32_bf16(false, a.v, false,
              *(const v16bf*)bbase, (short)0, acc0, false, false);
    acc1 = __builtin_amdgcn_wmma_f32_16x16x32_bf16(false, a.v, false,
              *(const v16bf*)(bbase + 512), (short)0, acc1, false, false);
  }

  // ---- assemble RBF output ----
  float rowsq = asum + __shfl_xor(asum, 16);  // full ||patch||^2 of row (lane&15)
  float wsq0  = lds_wsq[lane & 15];
  float wsq1  = lds_wsq[16 + (lane & 15)];
  int   nc0   = nbase + (lane & 15);
  int   nc1   = nc0 + 16;

  #pragma unroll
  for (int v = 0; v < 8; ++v) {
    int m   = mtile * 16 + v + 8 * hi;        // C layout: M = v + 8*(lane/16)
    int b2  = m / hw;
    int r2  = m - b2 * hw;
    int oy2 = r2 / Wo;
    int ox2 = r2 - oy2 * Wo;
    float xsq = __shfl(rowsq, v + 8 * hi);
    size_t ob = ((size_t)b2 * Cout) * hw + (size_t)oy2 * Wo + ox2;
    out[ob + (size_t)nc0 * hw] =
        f2bfu(__expf(-gamma * (xsq - 2.0f * acc0[v] + wsq0)));
    out[ob + (size_t)nc1 * hw] =
        f2bfu(__expf(-gamma * (xsq - 2.0f * acc1[v] + wsq1)));
  }
}

// ---------------------------------------------------------------------------
// 2x2 max-pool + ReLU + eval-mode BatchNorm (bf16 in, fp32 out).
// ---------------------------------------------------------------------------
__global__ __launch_bounds__(256) void pool_relu_bn(
    const unsigned short* __restrict__ in, float* __restrict__ out,
    const float* __restrict__ gg, const float* __restrict__ bb,
    const float* __restrict__ mm, const float* __restrict__ vv,
    int C, int Hin, int Win, int Hout, int Wout, int total)
{
  int idx = blockIdx.x * blockDim.x + threadIdx.x;
  if (idx >= total) return;
  int ox = idx % Wout;
  int t  = idx / Wout;
  int oy = t % Hout; t /= Hout;
  int c  = t % C;
  int b  = t / C;
  size_t base = (((size_t)b * C + c) * Hin + 2 * oy) * Win + 2 * ox;
  float p0 = bfu2f(in[base]),       p1 = bfu2f(in[base + 1]);
  float p2 = bfu2f(in[base + Win]), p3 = bfu2f(in[base + Win + 1]);
  float p = fmaxf(fmaxf(p0, p1), fmaxf(p2, p3));
  p = fmaxf(p, 0.f);                           // ReLU before BN (reference order)
  float scale = gg[c] * rsqrtf(vv[c] + EPS_);
  out[idx] = (p - mm[c]) * scale + bb[c];
}

// ---------------------------------------------------------------------------
// fc1 split-K: block = 256 threads = 8 waves = one (kslice, ntile) pair
// covering all 8 M-tiles. The 16x672 fp32 W tile is DMA'd into LDS by the
// Tensor Data Mover (one tensor_load_to_lds per block, TENSORcnt-tracked),
// then all 8 waves consume it from LDS. Partial 16x16 tiles are combined
// with global_atomic_add_f32. 42 slices x 16 ntiles = 672 blocks = 5376
// waves -> enough in-flight traffic for the ~29MB HBM stream.
// ---------------------------------------------------------------------------
#define FC1_SLICES     42
#define FC1_KSTEPS_PER 21                     // 42*21*32 = 28224 = K
#define FC1_SLICE_LEN  (FC1_KSTEPS_PER * 32)  // 672 floats

__global__ __launch_bounds__(256) void fc1_wmma_splitk(
    const float* __restrict__ A, const float* __restrict__ W,
    float* __restrict__ acc_out, int M, int N, int K)
{
  __shared__ float lds_w[16 * FC1_SLICE_LEN]; // 43,008 B
  (void)M;

  const int tid    = threadIdx.x;
  const int wave   = tid >> 5;
  const int lane   = tid & 31;
  const int ntile  = blockIdx.x & 15;
  const int kslice = blockIdx.x >> 4;
  const int mtile  = wave;                    // 8 waves cover all M tiles
  const int kcol0  = kslice * FC1_SLICE_LEN;

  const float* wtile = W + (size_t)(ntile * 16) * K + kcol0;

#if HAVE_TDM
  if (wave == 0) {
    tdm_load_2d_f32(wtile, (unsigned)(size_t)(void*)lds_w,
                    16, FC1_SLICE_LEN, K);
    __builtin_amdgcn_s_wait_tensorcnt(0);
  }
#else
  for (int e = tid; e < 16 * FC1_SLICE_LEN; e += 256) {
    int r = e / FC1_SLICE_LEN;
    int c = e - r * FC1_SLICE_LEN;
    lds_w[e] = wtile[(size_t)r * K + c];
  }
#endif
  __syncthreads();

  const int hi   = lane >> 4;
  const int mrow = mtile * 16 + (lane & 15);
  const float* arow = A + (size_t)mrow * K;
  v8f acc = {};

  for (int s = 0; s < FC1_KSTEPS_PER; ++s) {
    const int kb = kcol0 + (s << 5);
    AFrag a, b;
    // A: two contiguous 8-float chunks at kb+8*hi and kb+16+8*hi
    const float4* pa = (const float4*)(arow + kb + 8 * hi);
    float4 f0 = pa[0], f1 = pa[1], f2 = pa[4], f3 = pa[5];
    float af[16] = { f0.x, f0.y, f0.z, f0.w,  f1.x, f1.y, f1.z, f1.w,
                     f2.x, f2.y, f2.z, f2.w,  f3.x, f3.y, f3.z, f3.w };
    #pragma unroll
    for (int i = 0; i < 16; ++i) a.u[i] = f2bfu(af[i]);
    // B from LDS: lane%16 = N, element = K (table order)
    const int col = (s << 5) + 8 * hi;
    #pragma unroll
    for (int i = 0; i < 16; ++i)
      b.u[i] = f2bfu(lds_w[(lane & 15) * FC1_SLICE_LEN + col + i + 8 * (i >> 3)]);
    // prefetch the A stream ~4 ksteps ahead
    if (s + 4 < FC1_KSTEPS_PER)
      __builtin_prefetch(arow + kb + 128 + 8 * hi, 0, 1);
    acc = __builtin_amdgcn_wmma_f32_16x16x32_bf16(false, a.v, false, b.v,
                                                  (short)0, acc, false, false);
  }

  int ncol = ntile * 16 + (lane & 15);
  #pragma unroll
  for (int v = 0; v < 8; ++v) {
    int m = mtile * 16 + v + 8 * hi;
    atomicAdd(&acc_out[(size_t)m * N + ncol], acc[v]);
  }
}

__global__ __launch_bounds__(256) void zero_f32(float* __restrict__ p, int n) {
  int i = blockIdx.x * blockDim.x + threadIdx.x;
  if (i < n) p[i] = 0.f;
}

// ---------------------------------------------------------------------------
// fc2 (10 classes): applies fc1's bias+ReLU on the fly, then ReLU +
// log_softmax. One wave per batch row; shuffle reductions over the logits.
// ---------------------------------------------------------------------------
__global__ __launch_bounds__(32) void fc2_logsoftmax(
    const float* __restrict__ Aacc, const float* __restrict__ b1,
    const float* __restrict__ W, const float* __restrict__ b2,
    float* __restrict__ out, int K, int N)
{
  int b    = blockIdx.x;
  int lane = threadIdx.x;
  float val = -INFINITY;
  if (lane < N) {
    float s = b2[lane];
    const float* a = Aacc + (size_t)b * K;
    const float* w = W + (size_t)lane * K;
    for (int k = 0; k < K; ++k)
      s += fmaxf(a[k] + b1[k], 0.f) * w[k];
    val = fmaxf(s, 0.f);
  }
  float mx = val;
  #pragma unroll
  for (int off = 16; off > 0; off >>= 1) mx = fmaxf(mx, __shfl_xor(mx, off));
  float e = (lane < N) ? __expf(val - mx) : 0.f;
  float se = e;
  #pragma unroll
  for (int off = 16; off > 0; off >>= 1) se += __shfl_xor(se, off);
  if (lane < N) out[b * N + lane] = val - mx - __logf(se);
}

// ---------------------------------------------------------------------------
extern "C" void kernel_launch(void* const* d_in, const int* in_sizes, int n_in,
                              void* d_out, int out_size, void* d_ws, size_t ws_size,
                              hipStream_t stream) {
  (void)in_sizes; (void)n_in; (void)out_size; (void)ws_size;

  const float* x     = (const float*)d_in[0];
  const float* w1    = (const float*)d_in[1];
  const float* w2    = (const float*)d_in[2];
  const float* bn1_g = (const float*)d_in[3];
  const float* bn1_b = (const float*)d_in[4];
  const float* bn1_m = (const float*)d_in[5];
  const float* bn1_v = (const float*)d_in[6];
  const float* bn2_g = (const float*)d_in[7];
  const float* bn2_b = (const float*)d_in[8];
  const float* bn2_m = (const float*)d_in[9];
  const float* bn2_v = (const float*)d_in[10];
  const float* fc1_w = (const float*)d_in[11];
  const float* fc1_b = (const float*)d_in[12];
  const float* fc2_w = (const float*)d_in[13];
  const float* fc2_b = (const float*)d_in[14];
  float* outp = (float*)d_out;

  const int B = 128;
  const int C0 = 3,  H0 = 96;                                 // input
  const int C1 = 32, H1c = 92, H1p = 46, K1 = 75,  KS1 = 3;   // conv1
  const int C2 = 64, H2c = 42, H2p = 21, K2 = 800, KS2 = 25;  // conv2
  const int FLAT = C2 * H2p * H2p;                            // 28224
  const int L1 = 256, NCLS = 10;

  // scratch layout in d_ws
  char* ws = (char*)d_ws;
  size_t off = 0;
  auto take = [&](size_t bytes) {
    char* p = ws + off;
    off = (off + bytes + 255) & ~(size_t)255;
    return (void*)p;
  };
  unsigned short* c1out = (unsigned short*)take((size_t)B * C1 * H1c * H1c * 2);
  float*          p1    = (float*)         take((size_t)B * C1 * H1p * H1p * 4);
  unsigned short* c2out = (unsigned short*)take((size_t)B * C2 * H2c * H2c * 2);
  float*          p2    = (float*)         take((size_t)B * C2 * H2p * H2p * 4);
  float*          f1acc = (float*)         take((size_t)B * L1 * 4);

  // ---- conv1: M = 128*92*92 -> 67,712 M-tiles; N = 32 in one block ----
  {
    dim3 grid(67712 / 4, C1 / 32);
    gconv_wmma<<<grid, 128, 2 * KS1 * 512 * 2, stream>>>(
        x, w1, c1out, C0, H0, H0, C1, H1c, H1c, K1, KS1, GAMMA_);
  }
  // ---- pool1 + relu + bn1 ----
  {
    int total = B * C1 * H1p * H1p;
    pool_relu_bn<<<(total + 255) / 256, 256, 0, stream>>>(
        c1out, p1, bn1_g, bn1_b, bn1_m, bn1_v, C1, H1c, H1c, H1p, H1p, total);
  }
  // ---- conv2: M = 128*42*42 -> 14,112 M-tiles; N = 64 in two blocks ----
  {
    dim3 grid(14112 / 4, C2 / 32);
    gconv_wmma<<<grid, 128, 2 * KS2 * 512 * 2, stream>>>(
        p1, w2, c2out, C1, H1p, H1p, C2, H2c, H2c, K2, KS2, GAMMA_);
  }
  // ---- pool2 + relu + bn2 ----
  {
    int total = B * C2 * H2p * H2p;
    pool_relu_bn<<<(total + 255) / 256, 256, 0, stream>>>(
        c2out, p2, bn2_g, bn2_b, bn2_m, bn2_v, C2, H2c, H2c, H2p, H2p, total);
  }
  // ---- fc1 split-K: zero accumulator, then TDM-fed partial-tile blocks ----
  zero_f32<<<(B * L1 + 255) / 256, 256, 0, stream>>>(f1acc, B * L1);
  fc1_wmma_splitk<<<FC1_SLICES * (L1 / 16), 256, 0, stream>>>(
      p2, fc1_w, f1acc, B, L1, FLAT);
  // ---- fc2 (+fc1 bias/relu) + log_softmax ----
  fc2_logsoftmax<<<B, 32, 0, stream>>>(f1acc, fc1_b, fc2_w, fc2_b, outp, L1, NCLS);
}
